// Attention_15659450761345
// MI455X (gfx1250) — compile-verified, matrix operand order checked by reading, not run on hardware
//
#include <hip/hip_runtime.h>
#include <hip/hip_bf16.h>
#include <math.h>

typedef __attribute__((ext_vector_type(16))) __bf16 v16bf;
typedef __attribute__((ext_vector_type(8)))  float  v8f;

#define B_   2
#define S_   2048
#define D_   4096
#define NH_  32
#define NKV_ 8
#define HD_  128
#define NREP_ (NH_ / NKV_)

// ---------- helpers ----------

__device__ __forceinline__ unsigned short f2bf_u16(float f) {
  union { float f; unsigned u; } x; x.f = f;
  return (unsigned short)((x.u + 0x7FFFu + ((x.u >> 16) & 1u)) >> 16);  // RNE
}

__device__ __forceinline__ __bf16 f2bf(float f) {
  union { unsigned short s; __bf16 b; } y; y.s = f2bf_u16(f);
  return y.b;
}

__device__ __forceinline__ unsigned pack2bf(float lo, float hi) {
  return (unsigned)f2bf_u16(lo) | ((unsigned)f2bf_u16(hi) << 16);
}

// Fragment K index for register pair p (0..7), lane half g (0..1)
// per ISA 16-bit A-matrix 16x32 layout.
__device__ __forceinline__ int frag_k0(int p, int g) {
  return (p < 4) ? (g * 8 + 2 * p) : (16 + g * 8 + 2 * (p - 4));
}

// Inverse: k (0..31) -> (g, p); bf16 half within the pair is (k & 1).
__device__ __forceinline__ void k2frag(int k, int& g, int& p) {
  g = (k >> 3) & 1;
  p = ((k & 16) ? 4 : 0) + ((k & 7) >> 1);
}

// A fragment (16x32 bf16) gathered from row-major f32 (used once per wave for Q)
__device__ __forceinline__ v16bf load_frag_a_f32(const float* src, int ld) {
  int lane = threadIdx.x & 31;
  int m = lane & 15, g = lane >> 4;
  v16bf out;
#pragma unroll
  for (int p = 0; p < 8; ++p) {
    int k = frag_k0(p, g);
    out[2 * p]     = f2bf(src[m * ld + k]);
    out[2 * p + 1] = f2bf(src[m * ld + k + 1]);
  }
  return out;
}

// ---------- one-time layout conversion passes (f32 -> bf16 fragment-major) ----------

// A-layout: out[((t*(C/32)+kc)*32+lane)*8+p] packs X[row=t*16+(lane&15)][k], X[..][k+1]
__global__ __launch_bounds__(256) void conv_a(const float* __restrict__ X,
                                              unsigned* __restrict__ out,
                                              int c32shift, int C, int total) {
  int s = blockIdx.x * 256 + threadIdx.x;
  if (s >= total) return;
  int p = s & 7;
  int l = (s >> 3) & 31;
  int kc = (s >> 8) & ((1 << c32shift) - 1);
  int t = s >> (8 + c32shift);
  int row = t * 16 + (l & 15);
  int k = kc * 32 + frag_k0(p, l >> 4);
  const float* src = X + (size_t)row * C + k;
  out[s] = pack2bf(src[0], src[1]);
}

// B-layout: out[((c*(K/32)+kc)*32+lane)*8+p] packs W[k][n=c*16+(lane&15)], W[k+1][n]
__global__ __launch_bounds__(256) void conv_b(const float* __restrict__ W,
                                              unsigned* __restrict__ out,
                                              int k32shift, int N, int total) {
  int s = blockIdx.x * 256 + threadIdx.x;
  if (s >= total) return;
  int p = s & 7;
  int l = (s >> 3) & 31;
  int kc = (s >> 8) & ((1 << k32shift) - 1);
  int c = s >> (8 + k32shift);
  int n = c * 16 + (l & 15);
  int k = kc * 32 + frag_k0(p, l >> 4);
  out[s] = pack2bf(W[(size_t)k * N + n], W[(size_t)(k + 1) * N + n]);
}

// K-cache -> per-head A-layout: [b][kvh][t(S/16)][kc(4)][lane][8 pairs]
__global__ __launch_bounds__(256) void conv_ka(const float* __restrict__ xk,
                                               unsigned* __restrict__ out) {
  int s = blockIdx.x * 256 + threadIdx.x;   // total = B*NKV*(S/16)*4*256 = 2M
  int p = s & 7;
  int l = (s >> 3) & 31;
  int kc = (s >> 8) & 3;
  int t = (s >> 10) & (S_ / 16 - 1);
  int kvh = (s >> 17) & (NKV_ - 1);
  int b = s >> 20;
  int row = t * 16 + (l & 15);
  int k = kc * 32 + frag_k0(p, l >> 4);
  const float* src = xk + ((size_t)(b * S_ + row) * NKV_ + kvh) * HD_ + k;
  out[s] = pack2bf(src[0], src[1]);
}

// V-cache -> per-head B-layout: [b][kvh][kb(S/32)][c(8)][lane][8 pairs]
__global__ __launch_bounds__(256) void conv_vb(const float* __restrict__ xv,
                                               unsigned* __restrict__ out) {
  int s = blockIdx.x * 256 + threadIdx.x;   // total = B*NKV*(S/32)*8*256 = 2M
  int p = s & 7;
  int l = (s >> 3) & 31;
  int c = (s >> 8) & 7;
  int kb = (s >> 11) & (S_ / 32 - 1);
  int kvh = (s >> 17) & (NKV_ - 1);
  int b = s >> 20;
  int n = c * 16 + (l & 15);
  int row = kb * 32 + frag_k0(p, l >> 4);
  const float* src = xv + ((size_t)(b * S_ + row) * NKV_ + kvh) * HD_ + n;
  out[s] = pack2bf(src[0], src[NKV_ * HD_]);  // k and k+1 are adjacent rows
}

// ---------- GEMM on fragment-major bf16: C(MxN) = A(MxK)*B(KxN) ----------
// Block = 128x128 tile, 8 waves; fragments loaded straight from global (L2-resident).

__global__ __launch_bounds__(256) void gemm_frag(const v16bf* __restrict__ Abf,
                                                 const v16bf* __restrict__ Bbf,
                                                 float* __restrict__ C,
                                                 int N, int K) {
  const int tid = threadIdx.x;
  const int wave = tid >> 5;
  const int lane = tid & 31;
  const int KC = K >> 5;
  const int mt = blockIdx.y * 8 + (wave >> 2) * 4;  // 4 M-subtiles per wave
  const int nt = blockIdx.x * 8 + (wave & 3) * 2;   // 2 N-subtiles per wave

  const v16bf* ap[4];
  const v16bf* bp[2];
#pragma unroll
  for (int a = 0; a < 4; ++a) ap[a] = Abf + ((size_t)(mt + a) * KC) * 32 + lane;
#pragma unroll
  for (int bi = 0; bi < 2; ++bi) bp[bi] = Bbf + ((size_t)(nt + bi) * KC) * 32 + lane;

  v8f acc[4][2] = {};
  for (int kc = 0; kc < KC; ++kc) {
    v16bf af[4], bfr[2];
#pragma unroll
    for (int a = 0; a < 4; ++a) af[a] = ap[a][(size_t)kc * 32];
#pragma unroll
    for (int bi = 0; bi < 2; ++bi) bfr[bi] = bp[bi][(size_t)kc * 32];
#pragma unroll
    for (int a = 0; a < 4; ++a)
#pragma unroll
      for (int bi = 0; bi < 2; ++bi)
        acc[a][bi] = __builtin_amdgcn_wmma_f32_16x16x32_bf16(
            false, af[a], false, bfr[bi], (short)0, acc[a][bi], false, false);
  }

  const int nn = lane & 15, g = lane >> 4;
#pragma unroll
  for (int a = 0; a < 4; ++a)
#pragma unroll
    for (int bi = 0; bi < 2; ++bi)
#pragma unroll
      for (int r = 0; r < 8; ++r) {
        int row = (mt + a) * 16 + r + 8 * g;
        int col = (nt + bi) * 16 + nn;
        C[(size_t)row * N + col] = acc[a][bi][r];
      }
}

// ---------- RoPE (interleaved pairs) ----------

__global__ void rope_kernel(float* __restrict__ t, const float* __restrict__ cs,
                            const float* __restrict__ sn, int nh) {
  int idx = blockIdx.x * blockDim.x + threadIdx.x;
  int total = B_ * S_ * nh * (HD_ / 2);
  if (idx >= total) return;
  int hd2 = idx & 63;
  int h = (idx >> 6) % nh;
  int s = (idx / (64 * nh)) % S_;
  int b = idx / (64 * nh * S_);
  size_t base = ((size_t)(b * S_ + s) * nh + h) * HD_ + hd2 * 2;
  float c = cs[s * 64 + hd2], si = sn[s * 64 + hd2];
  float t0 = t[base], t1 = t[base + 1];
  t[base]     = t0 * c - t1 * si;
  t[base + 1] = t0 * si + t1 * c;
}

// ---------- Flash attention: 1 wave = 16 queries x 1 head ----------
// K/V fragments come pre-swizzled from global (L2-resident bf16 caches).

__global__ __launch_bounds__(32) void attn_kernel(const float* __restrict__ xq,
                                                  const v16bf* __restrict__ Kbf,
                                                  const v16bf* __restrict__ Vbf,
                                                  float* __restrict__ xo) {
  __shared__ v16bf Pf[32];  // P tile as one A-fragment (1KB)
  const int qt  = blockIdx.x;
  const int h   = blockIdx.y;
  const int b   = blockIdx.z;
  const int kvh = h / NREP_;
  const int lane = threadIdx.x & 31;
  const int nn = lane & 15, g = lane >> 4;

  __bf16* Pb = (__bf16*)Pf;

  // per-head fragment bases
  const v16bf* Kh = Kbf + ((size_t)(b * NKV_ + kvh) * (S_ / 16)) * 4 * 32;
  const v16bf* Vh = Vbf + ((size_t)(b * NKV_ + kvh) * (S_ / 32)) * 8 * 32;

  // preload Q tile (16x128) as 4 A-fragments
  const float* qbase = xq + ((size_t)(b * S_ + qt * 16) * NH_ + h) * HD_;
  v16bf qa[4];
#pragma unroll
  for (int kc = 0; kc < 4; ++kc) qa[kc] = load_frag_a_f32(qbase + kc * 32, NH_ * HD_);

  v8f acc[8] = {};
  float run_max[8], run_sum[8];
#pragma unroll
  for (int r = 0; r < 8; ++r) { run_max[r] = -1e30f; run_sum[r] = 0.f; }

  const float scale = 0.08838834764831845f;  // 1/sqrt(128)
  const int nkb = (qt * 16 + 47) >> 5;

  int g0, p0i, g1, p1i;
  k2frag(nn, g0, p0i);
  k2frag(nn + 16, g1, p1i);

  for (int jb = 0; jb < nkb; ++jb) {
    const int j0 = jb * 32;

    // ---- scores: Q(16x128) x K^T -> two 16x16 tiles (8 WMMAs) ----
    v8f s0 = {};
    v8f s1 = {};
    const v16bf* kf0 = Kh + ((size_t)(jb * 2) * 4) * 32 + lane;
    const v16bf* kf1 = kf0 + 4 * 32;
#pragma unroll
    for (int kc = 0; kc < 4; ++kc) {
      v16bf kb0 = kf0[kc * 32];
      v16bf kb1 = kf1[kc * 32];
      s0 = __builtin_amdgcn_wmma_f32_16x16x32_bf16(false, qa[kc], false, kb0, (short)0, s0, false, false);
      s1 = __builtin_amdgcn_wmma_f32_16x16x32_bf16(false, qa[kc], false, kb1, (short)0, s1, false, false);
    }

    // ---- mask + online softmax; D layout row m = r + 8*g, col = nn ----
#pragma unroll
    for (int r = 0; r < 8; ++r) {
      int qrow = qt * 16 + r + 8 * g;
      float e0 = s0[r] * scale + ((j0 + nn)      <= qrow ? 0.f : -1e9f);
      float e1 = s1[r] * scale + ((j0 + 16 + nn) <= qrow ? 0.f : -1e9f);
      float v = fmaxf(e0, e1);
#pragma unroll
      for (int off = 1; off < 16; off <<= 1) v = fmaxf(v, __shfl_xor(v, off, 32));
      float m_new = fmaxf(run_max[r], v);
      float corr = __expf(run_max[r] - m_new);
      float p0 = __expf(e0 - m_new);
      float p1 = __expf(e1 - m_new);
      float rs = p0 + p1;
#pragma unroll
      for (int off = 1; off < 16; off <<= 1) rs += __shfl_xor(rs, off, 32);
      run_sum[r] = run_sum[r] * corr + rs;
      run_max[r] = m_new;
#pragma unroll
      for (int c = 0; c < 8; ++c) acc[c][r] *= corr;
      // write P as bf16 directly in A-fragment layout
      int m = r + 8 * g;
      Pb[(m + 16 * g0) * 16 + 2 * p0i + (nn & 1)] = f2bf(p0);
      Pb[(m + 16 * g1) * 16 + 2 * p1i + (nn & 1)] = f2bf(p1);
    }
    __syncthreads();

    // ---- out += P(16x32) x V(32x128) (8 WMMAs) ----
    v16bf pa = Pf[lane];
    const v16bf* vf = Vh + ((size_t)jb * 8) * 32 + lane;
#pragma unroll
    for (int c = 0; c < 8; ++c) {
      v16bf vb = vf[c * 32];
      acc[c] = __builtin_amdgcn_wmma_f32_16x16x32_bf16(false, pa, false, vb, (short)0, acc[c], false, false);
    }
    __syncthreads();
  }

  // ---- normalize + store: xo[b, q, h, :] ----
#pragma unroll
  for (int r = 0; r < 8; ++r) {
    float inv = 1.0f / run_sum[r];
    int qrow = qt * 16 + r + 8 * g;
    size_t obase = ((size_t)(b * S_ + qrow) * NH_ + h) * HD_;
#pragma unroll
    for (int c = 0; c < 8; ++c)
      xo[obase + c * 16 + nn] = acc[c][r] * inv;
  }
}

// ---------- launch ----------

extern "C" void kernel_launch(void* const* d_in, const int* in_sizes, int n_in,
                              void* d_out, int out_size, void* d_ws, size_t ws_size,
                              hipStream_t stream) {
  (void)in_sizes; (void)n_in; (void)out_size; (void)ws_size;
  const float* x    = (const float*)d_in[0];
  const float* wq   = (const float*)d_in[1];
  const float* wk   = (const float*)d_in[2];
  const float* wv   = (const float*)d_in[3];
  const float* wo   = (const float*)d_in[4];
  const float* fcos = (const float*)d_in[7];
  const float* fsin = (const float*)d_in[8];
  float* out = (float*)d_out;

  char* ws = (char*)d_ws;
  size_t off = 0;
  unsigned* xA  = (unsigned*)(ws + off); off += (size_t)B_ * S_ * D_ * 2;          // bf16 A-layout (x, later xo)
  unsigned* wB  = (unsigned*)(ws + off); off += (size_t)D_ * D_ * 2;               // bf16 B-layout (reused per weight)
  float*    xq  = (float*)(ws + off);    off += (size_t)B_ * S_ * NH_  * HD_ * 4;
  float*    xk  = (float*)(ws + off);    off += (size_t)B_ * S_ * NKV_ * HD_ * 4;
  float*    xv  = (float*)(ws + off);    off += (size_t)B_ * S_ * NKV_ * HD_ * 4;
  float*    xo  = (float*)(ws + off);    off += (size_t)B_ * S_ * NH_  * HD_ * 4;
  unsigned* Kbf = (unsigned*)(ws + off); off += (size_t)B_ * NKV_ * S_ * HD_ * 2;
  unsigned* Vbf = (unsigned*)(ws + off); off += (size_t)B_ * NKV_ * S_ * HD_ * 2;

  const int M = B_ * S_;
  dim3 blk(256);

  // x -> bf16 A-layout (K = D, c32shift = 7)
  int totA = M * D_ / 2;
  conv_a<<<totA / 256, blk, 0, stream>>>(x, xA, 7, D_, totA);

  // Q projection
  int totWq = D_ * (NH_ * HD_) / 2;
  conv_b<<<totWq / 256, blk, 0, stream>>>(wq, wB, 7, NH_ * HD_, totWq);
  gemm_frag<<<dim3((NH_ * HD_) / 128, M / 128), blk, 0, stream>>>(
      (const v16bf*)xA, (const v16bf*)wB, xq, NH_ * HD_, D_);

  // K projection
  int totWk = D_ * (NKV_ * HD_) / 2;
  conv_b<<<totWk / 256, blk, 0, stream>>>(wk, wB, 7, NKV_ * HD_, totWk);
  gemm_frag<<<dim3((NKV_ * HD_) / 128, M / 128), blk, 0, stream>>>(
      (const v16bf*)xA, (const v16bf*)wB, xk, NKV_ * HD_, D_);

  // V projection
  conv_b<<<totWk / 256, blk, 0, stream>>>(wv, wB, 7, NKV_ * HD_, totWk);
  gemm_frag<<<dim3((NKV_ * HD_) / 128, M / 128), blk, 0, stream>>>(
      (const v16bf*)xA, (const v16bf*)wB, xv, NKV_ * HD_, D_);

  // RoPE
  int qtot = B_ * S_ * NH_  * (HD_ / 2);
  int ktot = B_ * S_ * NKV_ * (HD_ / 2);
  rope_kernel<<<(qtot + 255) / 256, 256, 0, stream>>>(xq, fcos, fsin, NH_);
  rope_kernel<<<(ktot + 255) / 256, 256, 0, stream>>>(xk, fcos, fsin, NKV_);

  // K/V -> per-head swizzled bf16 caches
  int totKV = B_ * NKV_ * S_ * HD_ / 2;
  conv_ka<<<totKV / 256, blk, 0, stream>>>(xk, Kbf);
  conv_vb<<<totKV / 256, blk, 0, stream>>>(xv, Vbf);

  // attention
  attn_kernel<<<dim3(S_ / 16, NH_, B_), dim3(32), 0, stream>>>(
      xq, (const v16bf*)Kbf, (const v16bf*)Vbf, xo);

  // output projection: xo -> bf16 A-layout (reuse xA), wo -> B-layout
  conv_a<<<totA / 256, blk, 0, stream>>>(xo, xA, 7, D_, totA);
  int totWo = (NH_ * HD_) * D_ / 2;
  conv_b<<<totWo / 256, blk, 0, stream>>>(wo, wB, 7, D_, totWo);
  gemm_frag<<<dim3(D_ / 128, M / 128), blk, 0, stream>>>(
      (const v16bf*)xA, (const v16bf*)wB, out, D_, D_);
}
